// SparseMoELayer_56702158242181
// MI455X (gfx1250) — matrix operand order repaired and checked by reading, not compiled, not run on hardware
//
#include <hip/hip_runtime.h>
#include <hip/hip_bf16.h>
#include <math.h>

typedef __bf16 bf16;
typedef __attribute__((ext_vector_type(16))) __bf16 v16bf;
typedef __attribute__((ext_vector_type(8)))  __bf16 bf16x8;
typedef __attribute__((ext_vector_type(8)))  float  v8f;
typedef int v4i __attribute__((ext_vector_type(4)));

#define NEXP   8
#define CDIM   768
#define HDIM   3072
#define NTOK   8192
#define TILE   128
#define KSTEP  32
#define MTILES 136                 // >= ceil((16384 + 7*128)/128)
#define MAXSLOTS (MTILES * TILE)   // 17408 padded slots

// ---------------- CDNA5 async global->LDS support (probe-guarded) ----------------
#if defined(__has_builtin)
#  if __has_builtin(__builtin_amdgcn_global_load_async_to_lds_b128)
#    define HAVE_ASYNC 1
#  else
#    define HAVE_ASYNC 0
#  endif
#else
#  define HAVE_ASYNC 0
#endif

#if HAVE_ASYNC
__device__ __forceinline__ void async_cp16(const bf16* g, bf16* l) {
    __builtin_amdgcn_global_load_async_to_lds_b128(
        (__attribute__((address_space(1))) v4i*)(g),
        (__attribute__((address_space(3))) v4i*)(l), 0, 0);
}
__device__ __forceinline__ void wait_async4() {
#if __has_builtin(__builtin_amdgcn_s_wait_asynccnt)
    __builtin_amdgcn_s_wait_asynccnt(4);
#else
    asm volatile("s_wait_asynccnt 0x4" ::: "memory");
#endif
}
__device__ __forceinline__ void wait_async0() {
#if __has_builtin(__builtin_amdgcn_s_wait_asynccnt)
    __builtin_amdgcn_s_wait_asynccnt(0);
#else
    asm volatile("s_wait_asynccnt 0x0" ::: "memory");
#endif
}
#endif

// ---------------------------------------------------------------- init
__global__ void k_init(int* __restrict__ rows, unsigned* __restrict__ counts,
                       unsigned* __restrict__ cursors) {
    int i = blockIdx.x * 256 + threadIdx.x;
    if (i < MAXSLOTS) rows[i] = -1;
    if (i < NEXP) { counts[i] = 0u; cursors[i] = 0u; }
}

// ---------------------------------------------------------------- fp32 -> bf16
__global__ void k_cvt(const float* __restrict__ in, bf16* __restrict__ out, int count) {
    int i = blockIdx.x * 256 + threadIdx.x;
    if (i < count) out[i] = (bf16)in[i];
}

// ------------------------------------------- transpose+convert: [E][R][S]f32 -> [E][S][R]bf16
__global__ __launch_bounds__(256) void k_tcvt(const float* __restrict__ in,
                                              bf16* __restrict__ out, int R, int S) {
    __shared__ float t[32][33];
    int e  = blockIdx.z;
    int s0 = blockIdx.x * 32, r0 = blockIdx.y * 32;
    const float* ip = in + (size_t)e * R * S;
    bf16* op = out + (size_t)e * R * S;
    int tx = threadIdx.x & 31, ty = threadIdx.x >> 5;   // 32 x 8
#pragma unroll
    for (int j = 0; j < 32; j += 8)
        t[ty + j][tx] = ip[(size_t)(r0 + ty + j) * S + s0 + tx];
    __syncthreads();
#pragma unroll
    for (int j = 0; j < 32; j += 8)
        op[(size_t)(s0 + ty + j) * R + r0 + tx] = (bf16)t[tx][ty + j];
}

// ---------------------------------------------------------------- router: softmax + top-2
__global__ __launch_bounds__(256) void k_router(const float* __restrict__ x,
                                                const float* __restrict__ Wr,
                                                float* __restrict__ tok_w,
                                                int* __restrict__ tok_e,
                                                unsigned* __restrict__ counts) {
    __shared__ float wr[CDIM * NEXP];                    // 24 KB
    for (int i = threadIdx.x; i < CDIM * NEXP; i += 256) wr[i] = Wr[i];
    __syncthreads();
    int n = blockIdx.x * 256 + threadIdx.x;              // NTOK % 256 == 0
    const float* xr = x + (size_t)n * CDIM;
    float acc[NEXP];
#pragma unroll
    for (int e = 0; e < NEXP; e++) acc[e] = 0.f;
    for (int c = 0; c < CDIM; c++) {
        float xv = xr[c];
#pragma unroll
        for (int e = 0; e < NEXP; e++) acc[e] = fmaf(xv, wr[c * NEXP + e], acc[e]);
    }
    float m = acc[0];
#pragma unroll
    for (int e = 1; e < NEXP; e++) m = fmaxf(m, acc[e]);
    float p[NEXP], s = 0.f;
#pragma unroll
    for (int e = 0; e < NEXP; e++) { p[e] = __expf(acc[e] - m); s += p[e]; }
    int i0 = 0;
#pragma unroll
    for (int e = 1; e < NEXP; e++) if (p[e] > p[i0]) i0 = e;
    int i1 = (i0 == 0) ? 1 : 0;
#pragma unroll
    for (int e = 0; e < NEXP; e++) if (e != i0 && p[e] > p[i1]) i1 = e;
    float p0 = p[i0] / s, p1 = p[i1] / s;
    float inv = 1.f / (p0 + p1 + 1e-9f);
    tok_e[2 * n] = i0; tok_e[2 * n + 1] = i1;
    tok_w[2 * n] = p0 * inv; tok_w[2 * n + 1] = p1 * inv;
    atomicAdd(&counts[i0], 1u);
    atomicAdd(&counts[i1], 1u);
}

// ---------------------------------------------------------------- 128-aligned offsets + utilization
__global__ void k_offsets(const unsigned* __restrict__ counts, int* __restrict__ padOff,
                          float* __restrict__ util_out) {
    if (threadIdx.x == 0) {
        int off = 0; unsigned tot = 0;
        for (int e = 0; e < NEXP; e++) {
            padOff[e] = off;
            off += (int)((counts[e] + TILE - 1) / TILE) * TILE;
            tot += counts[e];
        }
        padOff[NEXP] = off;
        for (int e = 0; e < NEXP; e++)
            util_out[e] = (float)counts[e] / ((float)tot + 1e-9f);
    }
}

// ---------------------------------------------------------------- scatter tokens into expert lists
__global__ void k_scatter(const int* __restrict__ tok_e, const int* __restrict__ padOff,
                          unsigned* __restrict__ cursors, int* __restrict__ rows,
                          int* __restrict__ slot_of) {
    int n = blockIdx.x * 256 + threadIdx.x;
    if (n >= NTOK) return;
#pragma unroll
    for (int k = 0; k < 2; k++) {
        int e = tok_e[2 * n + k];
        unsigned idx = atomicAdd(&cursors[e], 1u);
        int slot = padOff[e] + (int)idx;
        rows[slot] = n;
        slot_of[2 * n + k] = slot;
    }
}

// ======================= shared WMMA mainloop (macro to keep both kernels in sync) ==========
// Each thread owns 2 A-chunks and 2 B-chunks of 16B per K-step. Branchless addressing.
// Async path: double-buffered LDS + global_load_async_to_lds_b128 + s_wait_asynccnt.
// Fallback:   single LDS buffer + register double-buffering + weight prefetch.

#if HAVE_ASYNC
#define GEMM_MAINLOOP(NSTEPS, LDK)                                                        \
    {                                                                                     \
        async_cp16(ga0, &As[0][r0][c0]);                                                  \
        async_cp16(ga1, &As[0][r1][c1]);                                                  \
        async_cp16(gb0, &Bs[0][r0][c0]);                                                  \
        async_cp16(gb1, &Bs[0][r1][c1]);                                                  \
        for (int s = 0; s < (NSTEPS); s++) {                                              \
            int cur = s & 1;                                                              \
            if (s + 1 < (NSTEPS)) {                                                       \
                int ko = (s + 1) * KSTEP;                                                 \
                async_cp16(ga0 + ko, &As[cur ^ 1][r0][c0]);                               \
                async_cp16(ga1 + ko, &As[cur ^ 1][r1][c1]);                               \
                async_cp16(gb0 + ko, &Bs[cur ^ 1][r0][c0]);                               \
                async_cp16(gb1 + ko, &Bs[cur ^ 1][r1][c1]);                               \
                wait_async4();                                                            \
            } else {                                                                      \
                wait_async0();                                                            \
            }                                                                             \
            __syncthreads();                                                              \
            _Pragma("unroll")                                                             \
            for (int mi = 0; mi < 4; mi++) {                                              \
                v16bf a;                                                                  \
                const bf16* ap = &As[cur][wm * 64 + mi * 16 + l15][0];                    \
                ((bf16x8*)&a)[0] = *(const bf16x8*)(ap + 8 * half);                       \
                ((bf16x8*)&a)[1] = *(const bf16x8*)(ap + 16 + 8 * half);                  \
                _Pragma("unroll")                                                         \
                for (int ni = 0; ni < 2; ni++) {                                          \
                    v16bf b;                                                              \
                    const bf16* bp = &Bs[cur][wn * 32 + ni * 16 + l15][0];                \
                    ((bf16x8*)&b)[0] = *(const bf16x8*)(bp + 8 * half);                   \
                    ((bf16x8*)&b)[1] = *(const bf16x8*)(bp + 16 + 8 * half);              \
                    acc[mi][ni] = __builtin_amdgcn_wmma_f32_16x16x32_bf16(                \
                        false, a, false, b, (short)0, acc[mi][ni], false, false);         \
                }                                                                         \
            }                                                                             \
            __syncthreads();                                                              \
        }                                                                                 \
    }
#define LDS_DECL  __shared__ bf16 As[2][TILE][KSTEP]; __shared__ bf16 Bs[2][TILE][KSTEP];
#else
#define GEMM_MAINLOOP(NSTEPS, LDK)                                                        \
    {                                                                                     \
        bf16x8 ra0 = *(const bf16x8*)ga0, ra1 = *(const bf16x8*)ga1;                      \
        bf16x8 rb0 = *(const bf16x8*)gb0, rb1 = *(const bf16x8*)gb1;                      \
        for (int s = 0; s < (NSTEPS); s++) {                                              \
            __syncthreads();                                                              \
            *(bf16x8*)&As[r0][c0] = ra0;                                                  \
            *(bf16x8*)&As[r1][c1] = ra1;                                                  \
            *(bf16x8*)&Bs[r0][c0] = rb0;                                                  \
            *(bf16x8*)&Bs[r1][c1] = rb1;                                                  \
            if (s + 1 < (NSTEPS)) {                                                       \
                int ko = (s + 1) * KSTEP;                                                 \
                ra0 = *(const bf16x8*)(ga0 + ko);                                         \
                ra1 = *(const bf16x8*)(ga1 + ko);                                         \
                rb0 = *(const bf16x8*)(gb0 + ko);                                         \
                rb1 = *(const bf16x8*)(gb1 + ko);                                         \
                __builtin_prefetch(gb0 + ko + KSTEP, 0, 1);                               \
            }                                                                             \
            __syncthreads();                                                              \
            _Pragma("unroll")                                                             \
            for (int mi = 0; mi < 4; mi++) {                                              \
                v16bf a;                                                                  \
                const bf16* ap = &As[wm * 64 + mi * 16 + l15][0];                         \
                ((bf16x8*)&a)[0] = *(const bf16x8*)(ap + 8 * half);                       \
                ((bf16x8*)&a)[1] = *(const bf16x8*)(ap + 16 + 8 * half);                  \
                _Pragma("unroll")                                                         \
                for (int ni = 0; ni < 2; ni++) {                                          \
                    v16bf b;                                                              \
                    const bf16* bp = &Bs[wn * 32 + ni * 16 + l15][0];                     \
                    ((bf16x8*)&b)[0] = *(const bf16x8*)(bp + 8 * half);                   \
                    ((bf16x8*)&b)[1] = *(const bf16x8*)(bp + 16 + 8 * half);              \
                    acc[mi][ni] = __builtin_amdgcn_wmma_f32_16x16x32_bf16(                \
                        false, a, false, b, (short)0, acc[mi][ni], false, false);         \
                }                                                                         \
            }                                                                             \
        }                                                                                 \
    }
#define LDS_DECL  __shared__ bf16 As[TILE][KSTEP]; __shared__ bf16 Bs[TILE][KSTEP];
#endif

// ---------------------------------------------------------------- GEMM1: h = gelu(x @ W1 + b1), bf16 out
__global__ __launch_bounds__(256) void k_gemm1(const bf16* __restrict__ xb,
                                               const bf16* __restrict__ w1t,   // [E][H][C]
                                               const float* __restrict__ b1,
                                               const int* __restrict__ rows,
                                               const int* __restrict__ padOff,
                                               bf16* __restrict__ hbuf) {
    LDS_DECL
    int rowBase = blockIdx.x * TILE;
    int e = -1;
#pragma unroll
    for (int i = 0; i < NEXP; i++)
        if (rowBase >= padOff[i] && rowBase < padOff[i + 1]) e = i;
    if (e < 0) return;
    int colBase = blockIdx.y * TILE;
    const bf16* Wt = w1t + (size_t)e * HDIM * CDIM;
    int lane = threadIdx.x & 31, wave = threadIdx.x >> 5;
    int wm = wave >> 2, wn = wave & 3;          // 2 x 4 wave grid: 64x32 per wave
    int half = lane >> 4, l15 = lane & 15;

    // branchless gather descriptors (hoisted out of the K loop)
    int i0c = threadIdx.x, i1c = threadIdx.x + 256;
    int r0 = i0c >> 2, c0 = (i0c & 3) * 8;
    int r1 = i1c >> 2, c1 = (i1c & 3) * 8;
    int t0 = rows[rowBase + r0]; t0 = (t0 < 0) ? 0 : t0;   // padded rows read token 0: never consumed
    int t1 = rows[rowBase + r1]; t1 = (t1 < 0) ? 0 : t1;
    const bf16* ga0 = xb + (size_t)t0 * CDIM + c0;
    const bf16* ga1 = xb + (size_t)t1 * CDIM + c1;
    const bf16* gb0 = Wt + (size_t)(colBase + r0) * CDIM + c0;
    const bf16* gb1 = Wt + (size_t)(colBase + r1) * CDIM + c1;

    v8f acc[4][2] = {};
    GEMM_MAINLOOP(CDIM / KSTEP, CDIM)

#pragma unroll
    for (int mi = 0; mi < 4; mi++) {
#pragma unroll
        for (int ni = 0; ni < 2; ni++) {
            int col = colBase + wn * 32 + ni * 16 + l15;
            float bb = b1[(size_t)e * HDIM + col];
#pragma unroll
            for (int r = 0; r < 8; r++) {
                int row = rowBase + wm * 64 + mi * 16 + r + half * 8;
                float v = acc[mi][ni][r] + bb;
                float g = 0.5f * v * (1.0f + erff(v * 0.70710678118654752f));
                hbuf[(size_t)row * HDIM + col] = (bf16)g;
            }
        }
    }
}

// ---------------------------------------------------------------- GEMM2: y = h @ W2 + b2, fp32 out
__global__ __launch_bounds__(256) void k_gemm2(const bf16* __restrict__ hbuf,
                                               const bf16* __restrict__ w2t,   // [E][C][H]
                                               const float* __restrict__ b2,
                                               const int* __restrict__ padOff,
                                               float* __restrict__ ybuf) {
    LDS_DECL
    int rowBase = blockIdx.x * TILE;
    int e = -1;
#pragma unroll
    for (int i = 0; i < NEXP; i++)
        if (rowBase >= padOff[i] && rowBase < padOff[i + 1]) e = i;
    if (e < 0) return;
    int colBase = blockIdx.y * TILE;
    const bf16* Wt = w2t + (size_t)e * HDIM * CDIM;
    int lane = threadIdx.x & 31, wave = threadIdx.x >> 5;
    int wm = wave >> 2, wn = wave & 3;
    int half = lane >> 4, l15 = lane & 15;

    int i0c = threadIdx.x, i1c = threadIdx.x + 256;
    int r0 = i0c >> 2, c0 = (i0c & 3) * 8;
    int r1 = i1c >> 2, c1 = (i1c & 3) * 8;
    const bf16* ga0 = hbuf + (size_t)(rowBase + r0) * HDIM + c0;
    const bf16* ga1 = hbuf + (size_t)(rowBase + r1) * HDIM + c1;
    const bf16* gb0 = Wt + (size_t)(colBase + r0) * HDIM + c0;
    const bf16* gb1 = Wt + (size_t)(colBase + r1) * HDIM + c1;

    v8f acc[4][2] = {};
    GEMM_MAINLOOP(HDIM / KSTEP, HDIM)

#pragma unroll
    for (int mi = 0; mi < 4; mi++) {
#pragma unroll
        for (int ni = 0; ni < 2; ni++) {
            int col = colBase + wn * 32 + ni * 16 + l15;
            float bb = b2[(size_t)e * CDIM + col];
#pragma unroll
            for (int r = 0; r < 8; r++) {
                int row = rowBase + wm * 64 + mi * 16 + r + half * 8;
                ybuf[(size_t)row * CDIM + col] = acc[mi][ni][r] + bb;
            }
        }
    }
}

// ---------------------------------------------------------------- combine: out = w0*y[s0] + w1*y[s1]
__global__ void k_combine(const float* __restrict__ ybuf, const float* __restrict__ tok_w,
                          const int* __restrict__ slot_of, float* __restrict__ out) {
    int i = blockIdx.x * 256 + threadIdx.x;          // over NTOK * CDIM / 4
    if (i >= NTOK * (CDIM / 4)) return;
    int n = i / (CDIM / 4), c = i % (CDIM / 4);
    float w0 = tok_w[2 * n], w1 = tok_w[2 * n + 1];
    const float4* y0 = (const float4*)(ybuf + (size_t)slot_of[2 * n] * CDIM);
    const float4* y1 = (const float4*)(ybuf + (size_t)slot_of[2 * n + 1] * CDIM);
    float4 a = y0[c], b = y1[c], r;
    r.x = w0 * a.x + w1 * b.x;
    r.y = w0 * a.y + w1 * b.y;
    r.z = w0 * a.z + w1 * b.z;
    r.w = w0 * a.w + w1 * b.w;
    ((float4*)out)[i] = r;
}

// ---------------------------------------------------------------- host launcher
extern "C" void kernel_launch(void* const* d_in, const int* in_sizes, int n_in,
                              void* d_out, int out_size, void* d_ws, size_t ws_size,
                              hipStream_t stream) {
    const float* x  = (const float*)d_in[0];
    const float* Wr = (const float*)d_in[1];
    const float* W1 = (const float*)d_in[2];
    const float* b1 = (const float*)d_in[3];
    const float* W2 = (const float*)d_in[4];
    const float* b2 = (const float*)d_in[5];
    float* out = (float*)d_out;

    char* p = (char*)d_ws;
    auto alloc = [&](size_t bytes) -> void* {
        void* r = (void*)p;
        p += (bytes + 255) & ~(size_t)255;
        return r;
    };
    bf16*     xb      = (bf16*)alloc((size_t)NTOK * CDIM * sizeof(bf16));
    bf16*     w1t     = (bf16*)alloc((size_t)NEXP * HDIM * CDIM * sizeof(bf16));
    bf16*     w2t     = (bf16*)alloc((size_t)NEXP * HDIM * CDIM * sizeof(bf16));
    bf16*     hbuf    = (bf16*)alloc((size_t)MAXSLOTS * HDIM * sizeof(bf16));
    float*    ybuf    = (float*)alloc((size_t)MAXSLOTS * CDIM * sizeof(float));
    float*    tok_w   = (float*)alloc((size_t)2 * NTOK * sizeof(float));
    int*      tok_e   = (int*)alloc((size_t)2 * NTOK * sizeof(int));
    int*      slot_of = (int*)alloc((size_t)2 * NTOK * sizeof(int));
    int*      rows    = (int*)alloc((size_t)MAXSLOTS * sizeof(int));
    unsigned* counts  = (unsigned*)alloc(NEXP * sizeof(unsigned));
    unsigned* cursors = (unsigned*)alloc(NEXP * sizeof(unsigned));
    int*      padOff  = (int*)alloc((NEXP + 1) * sizeof(int));

    k_init<<<(MAXSLOTS + 255) / 256, 256, 0, stream>>>(rows, counts, cursors);
    k_cvt<<<(NTOK * CDIM) / 256, 256, 0, stream>>>(x, xb, NTOK * CDIM);
    k_tcvt<<<dim3(HDIM / 32, CDIM / 32, NEXP), 256, 0, stream>>>(W1, w1t, CDIM, HDIM);
    k_tcvt<<<dim3(CDIM / 32, HDIM / 32, NEXP), 256, 0, stream>>>(W2, w2t, HDIM, CDIM);
    k_router<<<NTOK / 256, 256, 0, stream>>>(x, Wr, tok_w, tok_e, counts);
    k_offsets<<<1, 32, 0, stream>>>(counts, padOff, out + (size_t)NTOK * CDIM);
    k_scatter<<<NTOK / 256, 256, 0, stream>>>(tok_e, padOff, cursors, rows, slot_of);
    k_gemm1<<<dim3(MTILES, HDIM / TILE), 256, 0, stream>>>(xb, w1t, b1, rows, padOff, hbuf);
    k_gemm2<<<dim3(MTILES, CDIM / TILE), 256, 0, stream>>>(hbuf, w2t, b2, padOff, ybuf);
    k_combine<<<(NTOK * CDIM) / 1024, 256, 0, stream>>>(ybuf, tok_w, slot_of, out);
}
